// pyg_GCNLayer_with_edge_9294309228638
// MI455X (gfx1250) — compile-verified, matrix operand order checked by reading, not run on hardware
//
#include <hip/hip_runtime.h>

#define N_NODES 50000
#define N_EDGES 800000
#define DIN     128
#define DOUT    128
#define EDIM    16
#define BN_EPS  1e-5f

typedef __attribute__((ext_vector_type(16))) _Float16 v16h;
typedef __attribute__((ext_vector_type(8)))  float    v8f;

// K index for half-slot h (0..15) within a 16-bit WMMA A/B fragment,
// per CDNA5 ISA §7.12.2 (lane-half g = lane>>4):
//   VGPR v holds K pair {2(v%4), 2(v%4)+1} in K-group (v/4)*16 + g*8
__device__ __forceinline__ int kmap16(int h, int g) {
  int v = h >> 1, p = h & 1;
  return ((v >> 2) << 4) | (g << 3) | ((v & 3) << 1) | p;
}

__device__ __forceinline__ v8f wmma_f16(v16h a, v16h b, v8f c) {
  return __builtin_amdgcn_wmma_f32_16x16x32_f16(false, a, false, b,
                                                (short)0, c, false, false);
}

// A fragment straight from global with f32->f16 convert (used only for the tiny eagg tile)
__device__ __forceinline__ v16h load_frag_A(const float* __restrict__ M,
                                            int row0, int ld, int k0, int kmax,
                                            int lane) {
  v16h a;
  int g = lane >> 4;
  const float* rp = M + (long long)(row0 + (lane & 15)) * ld;
#pragma unroll
  for (int h = 0; h < 16; ++h) {
    int k = k0 + kmap16(h, g);
    a[h] = (k < kmax) ? (_Float16)rp[k] : (_Float16)0.0f;
  }
  return a;
}

__global__ void gcn_zero_kernel(float* __restrict__ p, long long n) {
  long long i = (long long)blockIdx.x * blockDim.x + threadIdx.x;
  if (i < n) p[i] = 0.0f;
}

// Pack W_rel / W_res / W_edge into f16 WMMA B-fragment layout, once per launch.
// Fragment order: [n_tile][k_chunk][lane][16 halves], B[k][n] = W[n][k].
__global__ void gcn_pack_weights(const float* __restrict__ W_rel,
                                 const float* __restrict__ W_res,
                                 const float* __restrict__ W_edge,
                                 _Float16* __restrict__ Prel,
                                 _Float16* __restrict__ Pres,
                                 _Float16* __restrict__ Pedge) {
  int id = blockIdx.x * blockDim.x + threadIdx.x;   // 0..36863 exactly
  const float* W; _Float16* P; int ld, kmax, idx;
  if (id < 16384)      { W = W_rel;  P = Prel;  ld = DIN;  kmax = DIN;  idx = id; }
  else if (id < 32768) { W = W_res;  P = Pres;  ld = DIN;  kmax = DIN;  idx = id - 16384; }
  else                 { W = W_edge; P = Pedge; ld = EDIM; kmax = EDIM; idx = id - 32768; }
  int h    = idx & 15;
  int lane = (idx >> 4) & 31;
  int fc   = idx >> 9;                       // tile*nchunks + chunk
  int nchunks = (kmax == DIN) ? 4 : 1;
  int tile = fc / nchunks, chunk = fc % nchunks;
  int n = tile * 16 + (lane & 15);
  int k = chunk * 32 + kmap16(h, lane >> 4);
  P[idx] = (k < kmax) ? (_Float16)W[n * ld + k] : (_Float16)0.0f;
}

// One thread per (edge, out-channel). Scatter raw inputs; all targets L2-resident.
__global__ void gcn_edge_scatter(const int* __restrict__ ei,     // [2,E] (src row, then dst row)
                                 const float* __restrict__ feats,
                                 const float* __restrict__ eattr,
                                 float* __restrict__ aggF,       // [N,128] sum feats[src]
                                 float* __restrict__ eagg,       // [N,16]  sum edge_attr
                                 float* __restrict__ deg) {      // [N]
  long long gid = (long long)blockIdx.x * blockDim.x + threadIdx.x;
  int e = (int)(gid >> 7);
  int c = (int)(gid & 127);
  if (e >= N_EDGES) return;
  int s = ei[e];
  int d = ei[N_EDGES + e];
  atomicAdd(&aggF[(long long)d * DIN + c], feats[(long long)s * DIN + c]);
  if (c < EDIM) atomicAdd(&eagg[d * EDIM + c], eattr[e * EDIM + c]);
  if (c == 0)   atomicAdd(&deg[d], 1.0f);
}

// One block = 16 nodes; wave w computes output channels [16w,16w+16).
// A-tiles (aggF, feats rows) staged through LDS with a single f32->f16 pass;
// B-fragments come pre-packed from gcn_pack_weights.
__global__ void gcn_node_fused(const float* __restrict__ aggF,
                               const float* __restrict__ eagg,
                               const float* __restrict__ deg,
                               const float* __restrict__ feats,
                               const v16h* __restrict__ Brel,
                               const v16h* __restrict__ Bres,
                               const v16h* __restrict__ Bedge,
                               const float* __restrict__ b_rel,
                               const float* __restrict__ b_edge,
                               const float* __restrict__ b_res,
                               float* __restrict__ out,     // pre-BN h, [N,128]
                               float* __restrict__ sums,    // [128] sum h
                               float* __restrict__ sumsq) { // [128] sum h^2
  __shared__ v16h As[4 * 32];   // aggF tile, fragment layout [chunk][lane]
  __shared__ v16h Fs[4 * 32];   // feats tile
  int t  = threadIdx.x;
  int m0 = blockIdx.x * 16;

  // Stage + convert: idx = t*8+i keeps (chunk,lane,h-half) fixed per thread,
  // k runs over 8 consecutive values -> 32B coalesced global reads.
  _Float16* Ap = (_Float16*)As;
  _Float16* Fp = (_Float16*)Fs;
#pragma unroll
  for (int i = 0; i < 8; ++i) {
    int idx   = t * 8 + i;              // 0..2047
    int chunk = idx >> 9;
    int lane  = (idx >> 4) & 31;
    int h     = idx & 15;
    int k     = chunk * 32 + kmap16(h, lane >> 4);
    long long row = (long long)(m0 + (lane & 15)) * DIN;
    Ap[idx] = (_Float16)aggF[row + k];
    Fp[idx] = (_Float16)feats[row + k];
  }
  __syncthreads();

  int lane = t & 31;
  int wave = t >> 5;   // n-tile index, n0 = 16*wave
  v8f acc  = {};
  v8f accR = {};
#pragma unroll
  for (int c = 0; c < 4; ++c) {
    v16h a  = As[c * 32 + lane];
    v16h b  = Brel[(wave * 4 + c) * 32 + lane];
    acc  = wmma_f16(a, b, acc);
    v16h af = Fs[c * 32 + lane];
    v16h br = Bres[(wave * 4 + c) * 32 + lane];
    accR = wmma_f16(af, br, accR);
  }
  {
    // Edge-feature GEMM, K=16 zero-padded to 32 (tile is tiny: 16x16 f32)
    v16h ae = load_frag_A(eagg, m0, EDIM, 0, EDIM, lane);
    v16h be = Bedge[wave * 32 + lane];
    acc = wmma_f16(ae, be, acc);
  }

  // C/D layout: VGPR r, lanes 0-15 -> M=r, lanes 16-31 -> M=r+8; N = lane&15
  int n = wave * 16 + (lane & 15);
  float bsum = b_rel[n] + b_edge[n];
  float brn  = b_res[n];
  float s = 0.0f, s2 = 0.0f;
#pragma unroll
  for (int r = 0; r < 8; ++r) {
    int m = m0 + r + ((lane >> 4) << 3);
    float hv = acc[r] + deg[m] * bsum;
    hv = hv > 0.0f ? hv : 0.0f;
    float rv = accR[r] + brn;
    rv = rv > 0.0f ? rv : 0.0f;
    float o = hv + rv;
    out[(long long)m * DOUT + n] = o;
    s  += o;
    s2 += o * o;
  }
  // lanes L and L^16 hold the same channel n -> combine, lanes 0-15 commit
  s  += __shfl_xor(s, 16, 32);
  s2 += __shfl_xor(s2, 16, 32);
  if (lane < 16) {
    atomicAdd(&sums[n], s);
    atomicAdd(&sumsq[n], s2);
  }
}

__global__ void gcn_bn_finalize(float* __restrict__ out,
                                const float* __restrict__ sums,
                                const float* __restrict__ sumsq,
                                const float* __restrict__ gamma,
                                const float* __restrict__ beta) {
  int i = blockIdx.x * blockDim.x + threadIdx.x;
  if (i >= N_NODES * DOUT) return;
  int n = i & (DOUT - 1);
  const float invN = 1.0f / (float)N_NODES;
  float mean = sums[n] * invN;
  float var  = sumsq[n] * invN - mean * mean;
  out[i] = gamma[n] * (out[i] - mean) * rsqrtf(var + BN_EPS) + beta[n];
}

extern "C" void kernel_launch(void* const* d_in, const int* in_sizes, int n_in,
                              void* d_out, int out_size, void* d_ws, size_t ws_size,
                              hipStream_t stream) {
  const float* feats  = (const float*)d_in[0];
  const int*   ei     = (const int*)  d_in[1];
  const float* eattr  = (const float*)d_in[2];
  const float* W_rel  = (const float*)d_in[3];
  const float* b_rel  = (const float*)d_in[4];
  const float* W_edge = (const float*)d_in[5];
  const float* b_edge = (const float*)d_in[6];
  const float* W_res  = (const float*)d_in[7];
  const float* b_res  = (const float*)d_in[8];
  const float* gamma  = (const float*)d_in[9];
  const float* beta   = (const float*)d_in[10];
  float* out = (float*)d_out;

  // Workspace (floats): aggF[N*128] | eagg[N*16] | deg[N] | sums[128] | sumsq[128]
  // then (f16): Prel[16384] | Pres[16384] | Pedge[4096]   (~29.1 MB total)
  float* ws    = (float*)d_ws;
  float* aggF  = ws;
  float* eagg  = aggF + (long long)N_NODES * DIN;
  float* deg   = eagg + (long long)N_NODES * EDIM;
  float* sums  = deg + N_NODES;
  float* sumsq = sums + DOUT;
  _Float16* pk = (_Float16*)(sumsq + DOUT);    // 32B-aligned (offset 29,001,024 B)
  _Float16* Prel  = pk;
  _Float16* Pres  = pk + 16384;
  _Float16* Pedge = pk + 32768;

  long long zn = (long long)N_NODES * (DIN + EDIM + 1) + 2 * DOUT;
  gcn_zero_kernel<<<(int)((zn + 255) / 256), 256, 0, stream>>>(ws, zn);

  gcn_pack_weights<<<144, 256, 0, stream>>>(W_rel, W_res, W_edge, Prel, Pres, Pedge);

  long long ework = (long long)N_EDGES * DIN;
  gcn_edge_scatter<<<(int)((ework + 255) / 256), 256, 0, stream>>>(
      ei, feats, eattr, aggF, eagg, deg);

  gcn_node_fused<<<N_NODES / 16, 256, 0, stream>>>(
      aggF, eagg, deg, feats,
      (const v16h*)Prel, (const v16h*)Pres, (const v16h*)Pedge,
      b_rel, b_edge, b_res, out, sums, sumsq);

  int total = N_NODES * DOUT;
  gcn_bn_finalize<<<(total + 255) / 256, 256, 0, stream>>>(out, sums, sumsq, gamma, beta);
}
